// HopfieldNet_43997644981026
// MI455X (gfx1250) — matrix operand order repaired and checked by reading, not compiled, not run on hardware
//
#include <hip/hip_runtime.h>
#include <math.h>

typedef __attribute__((ext_vector_type(16))) __bf16 v16bf;
typedef __attribute__((ext_vector_type(8)))  float  v8f;

#define DLAT 4096            // latent dim (M and N of output)
#define NROW 1024            // rows of x (K of the GEMM)
#define NELEM (DLAT * NROW)  // 4,194,304 elements
#define NTRI  528            // 32*33/2 upper-triangular 128x128 blocks

// --------------------------------------------------------------------------
// Stage 1: deterministic two-stage mean reduction (no float atomics)
// --------------------------------------------------------------------------
__global__ __launch_bounds__(256) void hn_reduce1(const float* __restrict__ x,
                                                  float* __restrict__ partials) {
    __shared__ float s[256];
    int t = threadIdx.x;
    int base = blockIdx.x * 4096;           // 256 threads * 16 elems per block
    float sum = 0.f;
    #pragma unroll
    for (int i = 0; i < 16; ++i) sum += x[base + t + i * 256];
    s[t] = sum;
    __syncthreads();
    for (int st = 128; st > 0; st >>= 1) {
        if (t < st) s[t] += s[t + st];
        __syncthreads();
    }
    if (t == 0) partials[blockIdx.x] = s[0];
}

__global__ __launch_bounds__(256) void hn_reduce2(const float* __restrict__ partials,
                                                  float* __restrict__ rho) {
    __shared__ float s[256];
    int t = threadIdx.x;
    float sum = 0.f;
    #pragma unroll
    for (int i = 0; i < 4; ++i) sum += partials[t + i * 256];
    s[t] = sum;
    __syncthreads();
    for (int st = 128; st > 0; st >>= 1) {
        if (t < st) s[t] += s[t + st];
        __syncthreads();
    }
    if (t == 0) rho[0] = s[0] * (1.0f / (float)NELEM);
}

// --------------------------------------------------------------------------
// Stage 2: subtract rho, split fp32 -> bf16 hi/lo, and pre-swizzle into the
// exact CDNA5 WMMA fragment layouts (ISA 7.12.2) so the GEMM reads fragments
// as contiguous 32B-per-lane vectors.
//
// A = xc^T (16x32 MxK tiles):  lane<16: row=lane,     K = j<8 ? j    : j+8
//                              lane>=16: row=lane-16, K = j<8 ? j+8  : j+16
// B = xc   (32x16 KxN tiles):  lane<16: col=lane,     K = j<8 ? j    : j+8
//                              lane>=16: col=lane-16, K = j<8 ? j+8  : j+16
// Packed address: ((tile_mn*32 + tile_k)*32 + lane)*16 + j
// --------------------------------------------------------------------------
__global__ __launch_bounds__(256) void hn_pack(const float* __restrict__ x,
                                               const float* __restrict__ rhop,
                                               __bf16* __restrict__ Ah, __bf16* __restrict__ Al,
                                               __bf16* __restrict__ Bh, __bf16* __restrict__ Bl) {
    const float rho = rhop[0];
    int t    = blockIdx.x * 256 + threadIdx.x;   // 0..NELEM-1 packed position
    int j    = t & 15;          // element within lane's fragment
    int lane = (t >> 4) & 31;   // lane
    int tile = t >> 9;          // fragment tile: 0..8191
    int tk   = tile & 31;       // K tile (32 of them, K=32 each)
    int tm   = tile >> 5;       // M (or N) tile: 0..255
    int hiLane = lane >> 4;
    int lsub   = lane & 15;

    // A fragment: A[d,n] = xc[n,d]
    int ka = j + ((j >= 8) ? 8 : 0) + hiLane * 8;
    int d  = tm * 16 + lsub;
    int na = tk * 32 + ka;
    float av = x[na * DLAT + d] - rho;
    __bf16 avh = (__bf16)av;
    __bf16 avl = (__bf16)(av - (float)avh);
    Ah[t] = avh;  Al[t] = avl;

    // B fragment: B[n,e] = xc[n,e]
    int kb = j + ((j >= 8) ? 8 : 0) + hiLane * 8;
    int e  = tm * 16 + lsub;
    int nb = tk * 32 + kb;
    float bv = x[nb * DLAT + e] - rho;
    __bf16 bvh = (__bf16)bv;
    __bf16 bvl = (__bf16)(bv - (float)bvh);
    Bh[t] = bvh;  Bl[t] = bvl;
}

// --------------------------------------------------------------------------
// Stage 3: bf16x3 emulated-fp32 *SYRK* via v_wmma_f32_16x16x32_bf16.
// W is symmetric, so only the 528 upper-triangular 128x128 blocks are
// computed (1.94x less matrix math); mirror tiles are written via an LDS
// transpose so both halves store coalesced.
// Block = 256 threads = 8 wave32s; wave (wr,wc) owns 4x2 16x16 accumulators;
// 3 WMMAs per tile product (hi*hi + hi*lo + lo*hi) -> 24 WMMAs per K-step.
// --------------------------------------------------------------------------
__global__ __launch_bounds__(256) void hn_gemm(const __bf16* __restrict__ Ah,
                                               const __bf16* __restrict__ Al,
                                               const __bf16* __restrict__ Bh,
                                               const __bf16* __restrict__ Bl,
                                               float* __restrict__ out) {
    // Decode upper-triangular block pair (p <= q): l = q*(q+1)/2 + p
    int l = blockIdx.x;
    int q = (int)((sqrtf((float)(8 * l + 1)) - 1.0f) * 0.5f);
    while ((q + 1) * (q + 2) / 2 <= l) ++q;   // integer fixup (sqrt rounding)
    while (q * (q + 1) / 2 > l) --q;
    int p = l - q * (q + 1) / 2;              // 0 <= p <= q < 32

    int lane = threadIdx.x & 31;
    int wave = threadIdx.x >> 5;
    int wr = wave >> 2;                       // 0..1  (64-row slab)
    int wc = wave & 3;                        // 0..3  (32-col slab)
    int tdBase = p * 8 + wr * 4;              // 4 M-tiles (rows d)
    int teBase = q * 8 + wc * 2;              // 2 N-tiles (cols e)

    v8f acc[4][2] = {};

    #pragma unroll 1
    for (int tk = 0; tk < 32; ++tk) {
        v16bf ah[4], al[4], bh[2], bl[2];
        #pragma unroll
        for (int i = 0; i < 4; ++i) {
            int off = (((tdBase + i) * 32 + tk) * 32 + lane) * 16;
            ah[i] = *(const v16bf*)(Ah + off);
            al[i] = *(const v16bf*)(Al + off);
        }
        #pragma unroll
        for (int jj = 0; jj < 2; ++jj) {
            int off = (((teBase + jj) * 32 + tk) * 32 + lane) * 16;
            bh[jj] = *(const v16bf*)(Bh + off);
            bl[jj] = *(const v16bf*)(Bl + off);
        }
        // Operands are L2-resident (32 MB packed vs 192 MB L2): hint next step.
        if (tk + 1 < 32) {
            int offA = ((tdBase * 32 + (tk + 1)) * 32 + lane) * 16;
            int offB = ((teBase * 32 + (tk + 1)) * 32 + lane) * 16;
            __builtin_prefetch(Ah + offA, 0, 0);
            __builtin_prefetch(Bh + offB, 0, 0);
        }
        #pragma unroll
        for (int i = 0; i < 4; ++i) {
            #pragma unroll
            for (int jj = 0; jj < 2; ++jj) {
                acc[i][jj] = __builtin_amdgcn_wmma_f32_16x16x32_bf16(
                    false, ah[i], false, bh[jj], (short)0, acc[i][jj], false, false);
                acc[i][jj] = __builtin_amdgcn_wmma_f32_16x16x32_bf16(
                    false, ah[i], false, bl[jj], (short)0, acc[i][jj], false, false);
                acc[i][jj] = __builtin_amdgcn_wmma_f32_16x16x32_bf16(
                    false, al[i], false, bh[jj], (short)0, acc[i][jj], false, false);
            }
        }
    }

    // Epilogue. C/D layout: VGPR v -> row (v + 8*(lane>=16)), lane&15 -> col.
    // tb[wave][tile][row][col], col-padded to 17 so the transposed read
    // (lane stride 17 dwords, coprime with 64 banks) is conflict-free.
    __shared__ float tb[8][8][16][17];
    const float scale = 1.0f / 1024.0f;
    int colL   = lane & 15;
    int rowAdd = (lane >> 4) * 8;

    #pragma unroll
    for (int i = 0; i < 4; ++i) {
        int d0 = (tdBase + i) * 16;
        #pragma unroll
        for (int jj = 0; jj < 2; ++jj) {
            int e0 = (teBase + jj) * 16;
            int e  = e0 + colL;
            int ti = i * 2 + jj;
            #pragma unroll
            for (int v = 0; v < 8; ++v) {
                int d = d0 + rowAdd + v;
                float val = acc[i][jj][v] * scale;
                if (d == e) val = 0.f;
                out[(size_t)d * DLAT + e] = val;       // coalesced (16 lanes/row)
                tb[wave][ti][rowAdd + v][colL] = val;  // stage for mirror
            }
        }
    }

    if (p != q) {                                      // uniform per block
        __syncthreads();
        #pragma unroll
        for (int i = 0; i < 4; ++i) {
            int d0 = (tdBase + i) * 16;
            #pragma unroll
            for (int jj = 0; jj < 2; ++jj) {
                int e0 = (teBase + jj) * 16;
                int ti = i * 2 + jj;
                #pragma unroll
                for (int v = 0; v < 8; ++v) {
                    int ec = rowAdd + v;
                    float val = tb[wave][ti][colL][ec];            // transposed read
                    out[(size_t)(e0 + ec) * DLAT + (d0 + colL)] = val; // coalesced
                }
            }
        }
    }
}

// --------------------------------------------------------------------------
extern "C" void kernel_launch(void* const* d_in, const int* in_sizes, int n_in,
                              void* d_out, int out_size, void* d_ws, size_t ws_size,
                              hipStream_t stream) {
    const float* x  = (const float*)d_in[0];
    float* out      = (float*)d_out;
    char*  ws       = (char*)d_ws;

    float* rho      = (float*)ws;                 // 1 float
    float* partials = (float*)(ws + 256);         // 1024 floats
    const size_t PACK_BYTES = (size_t)NELEM * 2;  // 8 MiB per packed buffer
    __bf16* Ah = (__bf16*)(ws + 8192);
    __bf16* Al = (__bf16*)(ws + 8192 + 1 * PACK_BYTES);
    __bf16* Bh = (__bf16*)(ws + 8192 + 2 * PACK_BYTES);
    __bf16* Bl = (__bf16*)(ws + 8192 + 3 * PACK_BYTES);

    hn_reduce1<<<NELEM / 4096, 256, 0, stream>>>(x, partials);
    hn_reduce2<<<1, 256, 0, stream>>>(partials, rho);
    hn_pack<<<NELEM / 256, 256, 0, stream>>>(x, rho, Ah, Al, Bh, Bl);

    hn_gemm<<<NTRI, 256, 0, stream>>>(Ah, Al, Bh, Bl, out);  // triangular grid
}
